// SRONET_88304527606460
// MI455X (gfx1250) — matrix operand order repaired and checked by reading, not compiled, not run on hardware
//
#include <hip/hip_runtime.h>
#include <math.h>

// ---------------------------------------------------------------------------
// Types
// ---------------------------------------------------------------------------
typedef __attribute__((ext_vector_type(16))) __bf16 v16bf;
typedef __attribute__((ext_vector_type(8)))  float  v8f;
typedef unsigned int u32x4 __attribute__((ext_vector_type(4)));
typedef int          i32x4 __attribute__((ext_vector_type(4)));
typedef int          i32x8 __attribute__((ext_vector_type(8)));

struct alignas(16) U4 { unsigned int a, b, c, d; };
union Frag { U4 q[2]; v16bf v; };

__device__ __forceinline__ unsigned short f2bf(float f) {
    unsigned int u = __float_as_uint(f);
    return (unsigned short)((u + 0x7FFFu + ((u >> 16) & 1u)) >> 16);   // RNE
}
__device__ __forceinline__ float bf2f(unsigned short h) {
    return __uint_as_float(((unsigned int)h) << 16);
}

// Problem constants
#define NTOK   131072      // B(2) * 256 * 256
#define HQ     256
#define HLR    64
#define ENCC   64
#define WIDTHC 256
#define QKVC   768
#define KPAD   288         // 266 padded to multiple of 32

#if __has_builtin(__builtin_amdgcn_tensor_load_to_lds)
#define HAVE_TDM 1
#else
#define HAVE_TDM 0
#endif

// ---------------------------------------------------------------------------
// Weight fp32 [N][Ks] -> bf16 [N][Kd] (zero-padded K)
// ---------------------------------------------------------------------------
__global__ void k_cvt_w(const float* __restrict__ src, unsigned short* __restrict__ dst,
                        int N, int Ks, int Kd) {
    int t = blockIdx.x * blockDim.x + threadIdx.x;
    if (t >= N * Kd) return;
    int n = t / Kd, k = t - n * Kd;
    dst[t] = (k < Ks) ? f2bf(src[n * Ks + k]) : (unsigned short)0;
}

// ---------------------------------------------------------------------------
// Encoder conv3x3 SAME: inp [2][2][64][64] -> feat [2][64][64][64] fp32
// ---------------------------------------------------------------------------
__global__ void k_enc(const float* __restrict__ inp, const float* __restrict__ w,
                      const float* __restrict__ bias, float* __restrict__ feat) {
    int t = blockIdx.x * blockDim.x + threadIdx.x;
    if (t >= 2 * ENCC * HLR * HLR) return;
    int x = t & 63, y = (t >> 6) & 63, o = (t >> 12) & 63, b = t >> 18;
    float acc = bias[o];
    #pragma unroll
    for (int ci = 0; ci < 2; ci++)
        for (int ky = 0; ky < 3; ky++) {
            int yy = y + ky - 1;
            if (yy < 0 || yy > 63) continue;
            for (int kx = 0; kx < 3; kx++) {
                int xx = x + kx - 1;
                if (xx < 0 || xx > 63) continue;
                acc += inp[((b * 2 + ci) * HLR + yy) * HLR + xx] *
                       w[((o * 2 + ci) * 3 + ky) * 3 + kx];
            }
        }
    feat[t] = acc;
}

// ---------------------------------------------------------------------------
// Grid assembly (4 shifted nearest samples + rel coords + areas + rel_cell)
// ---------------------------------------------------------------------------
__global__ void k_build_grid(const float* __restrict__ feat, const float* __restrict__ coord,
                             const float* __restrict__ cell, unsigned short* __restrict__ grid) {
    int t = blockIdx.x * blockDim.x + threadIdx.x;
    if (t >= NTOK) return;
    int b = t >> 16, qy = (t >> 8) & 255, qx = t & 255;
    float gy = coord[((size_t)(b * HQ + qy) * HQ + qx) * 2 + 0];
    float gx = coord[((size_t)(b * HQ + qy) * HQ + qx) * 2 + 1];
    const float rxy = 1.0f / 64.0f;
    float area[4], relv[4][2];
    int   idx[4];
    int i = 0;
    #pragma unroll
    for (int vx = -1; vx <= 1; vx += 2)
        #pragma unroll
        for (int vy = -1; vy <= 1; vy += 2, i++) {
            float cy = fminf(fmaxf(gy + vx * rxy + 1e-6f, -1.f + 1e-6f), 1.f - 1e-6f);
            float cx = fminf(fmaxf(gx + vy * rxy + 1e-6f, -1.f + 1e-6f), 1.f - 1e-6f);
            float fx = ((cx + 1.f) * 64.f - 1.f) * 0.5f;
            float fy = ((cy + 1.f) * 64.f - 1.f) * 0.5f;
            int ix = (int)fminf(fmaxf(rintf(fx), 0.f), 63.f);
            int iy = (int)fminf(fmaxf(rintf(fy), 0.f), 63.f);
            float oy = -1.f + (2.f * iy + 1.f) / 64.f;
            float ox = -1.f + (2.f * ix + 1.f) / 64.f;
            float r0 = (gy - oy) * 64.f, r1 = (gx - ox) * 64.f;
            relv[i][0] = r0; relv[i][1] = r1;
            area[i] = fabsf(r0 * r1) + 1e-9f;
            idx[i] = iy * 64 + ix;
        }
    float tot = area[0] + area[1] + area[2] + area[3];
    unsigned short* g = grid + (size_t)t * KPAD;
    #pragma unroll
    for (i = 0; i < 4; i++) { g[2 * i] = f2bf(relv[i][0]); g[2 * i + 1] = f2bf(relv[i][1]); }
    const float* fb = feat + (size_t)b * ENCC * 4096;
    for (i = 0; i < 4; i++) {
        float wgt = area[3 - i] / tot;
        int p = idx[i];
        for (int c = 0; c < ENCC; c++)
            g[8 + i * ENCC + c] = f2bf(fb[c * 4096 + p] * wgt);
    }
    g[264] = f2bf(cell[b * 2 + 0] * 64.f);
    g[265] = f2bf(cell[b * 2 + 1] * 64.f);
    #pragma unroll
    for (int c = 266; c < KPAD; c++) g[c] = 0;
}

// ---------------------------------------------------------------------------
// Double-buffered async-copy WMMA GEMM:
//   out[M][N] = A[M][K] * W[N][K]^T + bias (+gelu | +residual)
// Tile 128x128, K-step 32. 8 waves, each 32(M) x 64(N) = 2x4 16x16 frags.
// A staged via global_load_async_to_lds_b128 (ASYNCcnt);
// B (weight tile) staged via one TDM descriptor per block (TENSORcnt).
// ---------------------------------------------------------------------------
#define TM 128
#define TN 128
#define TK 32
#define LDA 40    // ushort row stride (80B: 64B data + 16B TDM pad)
#define LDB 40

template <int FUSE>   // 0 = none, 1 = gelu, 2 = residual add
__global__ __launch_bounds__(256)
void k_gemm_bf16(const unsigned short* __restrict__ A,
                 const unsigned short* __restrict__ W,     // [N][K] bf16
                 const float* __restrict__ bias,
                 const unsigned short* __restrict__ res,   // [M][N] bf16 or null
                 unsigned short* __restrict__ out,         // [M][N] bf16
                 int M, int N, int K) {
    __shared__ alignas(16) unsigned short As[2][TM * LDA];
    __shared__ alignas(16) unsigned short Bs[2][TN * LDB];

    const int tid  = threadIdx.x;
    const int lane = tid & 31;
    const int wave = tid >> 5;
    const int wm = wave >> 1;            // 0..3  (M sub-tile, 32 rows)
    const int wn = wave & 1;             // 0..1  (N sub-tile, 64 cols)
    const int mBase = blockIdx.y * TM;
    const int nBase = blockIdx.x * TN;
    const int ml = lane & 15, kh = lane >> 4;

    // stage-copy assignments: 128 rows x 2 x 32B for both A and B
    const int cRow = tid & 127, cSeg = tid >> 7;   // cSeg: which 32B half-row

    const unsigned ldsA = (unsigned)(size_t)(&As[0][0]) +
                          (unsigned)(cRow * LDA + cSeg * 16) * 2;
    const unsigned ldsB0 = (unsigned)(size_t)(&Bs[0][0]);
    const unsigned ldsB = ldsB0 + (unsigned)(cRow * LDB + cSeg * 16) * 2;

    v8f acc[2][4];
    #pragma unroll
    for (int i = 0; i < 2; i++)
        #pragma unroll
        for (int j = 0; j < 4; j++)
            acc[i][j] = (v8f){0.f, 0.f, 0.f, 0.f, 0.f, 0.f, 0.f, 0.f};

    auto issue_stage = [&](int k0, int buf) {
        // ---- A tile: per-lane async global->LDS (2 x b128 per thread) ----
        unsigned ad = ldsA + (unsigned)buf * (TM * LDA * 2);
        unsigned long long ga =
            (unsigned long long)(A + (size_t)(mBase + cRow) * K + k0 + cSeg * 16);
        asm volatile("global_load_async_to_lds_b128 %0, %1, off"
                     :: "v"(ad), "v"(ga) : "memory");
        asm volatile("global_load_async_to_lds_b128 %0, %1, off"
                     :: "v"(ad + 16u), "v"(ga + 16ull) : "memory");
#if HAVE_TDM
        // ---- B tile: one Tensor-Data-Mover descriptor for the whole block ----
        if (tid == 0) {
            unsigned long long gw = (unsigned long long)(W + (size_t)nBase * K + k0);
            unsigned bdst = ldsB0 + (unsigned)buf * (TN * LDB * 2);
            u32x4 g0;
            g0[0] = 1u;                                            // count=1, user mode
            g0[1] = bdst;                                          // lds_addr (bytes)
            g0[2] = (unsigned)(gw & 0xffffffffull);                // global_addr[31:0]
            g0[3] = (unsigned)((gw >> 32) & 0x01ffffffull) | (2u << 30);  // [56:32]|type=2
            i32x8 g1;
            g1[0] = (int)((1u << 16)      // data_size = 2 bytes
                        | (1u << 20)      // pad_enable
                        | (3u << 22)      // pad_interval: 16 DWORDs (= 64B row)
                        | (3u << 25));    // pad_amount: 4 DWORDs (= 16B)
            g1[1] = (int)(((unsigned)K & 0xffffu) << 16);          // tensor_dim0 lo16
            g1[2] = (int)((((unsigned)K >> 16) & 0xffffu) |
                          (((unsigned)N & 0xffffu) << 16));        // dim0 hi | dim1 lo
            g1[3] = (int)((((unsigned)N >> 16) & 0xffffu) |
                          ((unsigned)TK << 16));                   // dim1 hi | tile_dim0=32
            g1[4] = (int)TN;                                       // tile_dim1=128, tile_dim2=0
            g1[5] = (int)K;                                        // tensor_dim0_stride lo32
            g1[6] = 0;                                             // stride hi | dim1_stride lo
            g1[7] = 0;
            i32x4 z = {0, 0, 0, 0};
#if __has_include(<hip/amd_detail/amd_gfx1250_TDM.h>)
            i32x8 z8 = {0, 0, 0, 0, 0, 0, 0, 0};
            __builtin_amdgcn_tensor_load_to_lds(g0, g1, z, z, z8, 0);   // clang-23 form
#else
            __builtin_amdgcn_tensor_load_to_lds(g0, g1, z, z, 0);       // ROCm 7.2 form
#endif
        }
#else
        // ---- B tile fallback: per-lane async copies ----
        unsigned bd = ldsB + (unsigned)buf * (TN * LDB * 2);
        unsigned long long gb =
            (unsigned long long)(W + (size_t)(nBase + cRow) * K + k0 + cSeg * 16);
        asm volatile("global_load_async_to_lds_b128 %0, %1, off"
                     :: "v"(bd), "v"(gb) : "memory");
        asm volatile("global_load_async_to_lds_b128 %0, %1, off"
                     :: "v"(bd + 16u), "v"(gb + 16ull) : "memory");
#endif
    };

#if HAVE_TDM
    #define STAGE_WAIT_NEXT() do { asm volatile("s_wait_asynccnt 2" ::: "memory"); \
                                   __builtin_amdgcn_s_wait_tensorcnt(1); } while (0)
    #define STAGE_WAIT_LAST() do { asm volatile("s_wait_asynccnt 0" ::: "memory"); \
                                   __builtin_amdgcn_s_wait_tensorcnt(0); } while (0)
#else
    #define STAGE_WAIT_NEXT() asm volatile("s_wait_asynccnt 4" ::: "memory")
    #define STAGE_WAIT_LAST() asm volatile("s_wait_asynccnt 0" ::: "memory")
#endif

    const int nsteps = K / TK;
    issue_stage(0, 0);

    for (int s = 0; s < nsteps; s++) {
        const int buf = s & 1;
        if (s + 1 < nsteps) { issue_stage((s + 1) * TK, buf ^ 1); STAGE_WAIT_NEXT(); }
        else                { STAGE_WAIT_LAST(); }
        __syncthreads();

        // ---- fragments (ISA 7.12.2 16-bit A / B lane layouts) ----
        Frag a[2], b[4];
        #pragma unroll
        for (int fm = 0; fm < 2; fm++) {
            const unsigned short* p = &As[buf][(wm * 32 + fm * 16 + ml) * LDA];
            a[fm].q[0] = *(const U4*)(p + kh * 8);        // K 0..7   (8..15)
            a[fm].q[1] = *(const U4*)(p + 16 + kh * 8);   // K 16..23 (24..31)
        }
        #pragma unroll
        for (int fn = 0; fn < 4; fn++) {
            const unsigned short* p = &Bs[buf][(wn * 64 + fn * 16 + ml) * LDB];
            b[fn].q[0] = *(const U4*)(p + kh * 8);
            b[fn].q[1] = *(const U4*)(p + 16 + kh * 8);
        }
        #pragma unroll
        for (int fm = 0; fm < 2; fm++)
            #pragma unroll
            for (int fn = 0; fn < 4; fn++)
                acc[fm][fn] = __builtin_amdgcn_wmma_f32_16x16x32_bf16(
                    false, a[fm].v, false, b[fn].v, (short)0, acc[fm][fn], false, false);
        __syncthreads();
    }

    // ---- epilogue: bias (+gelu | +residual), store bf16 ----
    #pragma unroll
    for (int fm = 0; fm < 2; fm++)
        #pragma unroll
        for (int fn = 0; fn < 4; fn++) {
            int n = nBase + wn * 64 + fn * 16 + ml;
            float bv = bias[n];
            #pragma unroll
            for (int r = 0; r < 8; r++) {
                int m = mBase + wm * 32 + fm * 16 + kh * 8 + r;
                float v = acc[fm][fn][r] + bv;
                if (FUSE == 1) v = 0.5f * v * (1.f + erff(v * 0.70710678118f));
                size_t off = (size_t)m * N + n;
                if (FUSE == 2) v += bf2f(res[off]);
                out[off] = f2bf(v);
            }
        }
}

// ---------------------------------------------------------------------------
// LayerNorm (ddof=1, /(std+eps)) on k and v slices of qkv, in place
// ---------------------------------------------------------------------------
__device__ __forceinline__ void ln16(unsigned short* p, const float* w, const float* b) {
    float x[16], s = 0.f;
    #pragma unroll
    for (int j = 0; j < 16; j++) { x[j] = bf2f(p[j]); s += x[j]; }
    float mean = s * (1.f / 16.f), var = 0.f;
    #pragma unroll
    for (int j = 0; j < 16; j++) { float d = x[j] - mean; var += d * d; }
    float inv = 1.f / (sqrtf(var * (1.f / 15.f)) + 1e-5f);
    #pragma unroll
    for (int j = 0; j < 16; j++) p[j] = f2bf(w[j] * (x[j] - mean) * inv + b[j]);
}

__global__ void k_ln_kv(unsigned short* __restrict__ qkv,
                        const float* __restrict__ klnw, const float* __restrict__ klnb,
                        const float* __restrict__ vlnw, const float* __restrict__ vlnb) {
    int t = blockIdx.x * blockDim.x + threadIdx.x;
    if (t >= NTOK * 16) return;
    int tok = t >> 4, hh = t & 15;
    unsigned short* base = qkv + (size_t)tok * QKVC + hh * 48;
    ln16(base + 16, klnw + hh * 16, klnb + hh * 16);
    ln16(base + 32, vlnw + hh * 16, vlnb + hh * 16);
}

// ---------------------------------------------------------------------------
// kv[b][h][d][e] = (1/65536) * sum_n khat[n,d] * vhat[n,e]   (LDS-staged)
// ---------------------------------------------------------------------------
__global__ __launch_bounds__(256) void k_kv(const unsigned short* __restrict__ qkv,
                                            float* __restrict__ kv) {
    int bh = blockIdx.x;                 // 0..31 = b*16+h
    int b = bh >> 4, hh = bh & 15;
    int d = threadIdx.x >> 4, e = threadIdx.x & 15;
    __shared__ float sk[128][16];
    __shared__ float sv[128][17];
    float acc = 0.f;
    const unsigned short* base = qkv + (size_t)b * 65536 * QKVC + hh * 48;
    for (int t0 = 0; t0 < 65536; t0 += 128) {
        int r = threadIdx.x >> 1, half = threadIdx.x & 1;
        const unsigned short* p = base + (size_t)(t0 + r) * QKVC + 16 + half * 16;
        float* dst = half ? sv[r] : sk[r];
        #pragma unroll
        for (int j = 0; j < 16; j++) dst[j] = bf2f(p[j]);
        __syncthreads();
        #pragma unroll 8
        for (int i = 0; i < 128; i++) acc = fmaf(sk[i][d], sv[i][e], acc);
        __syncthreads();
    }
    kv[((size_t)bh * 16 + d) * 16 + e] = acc * (1.f / 65536.f);
}

// ---------------------------------------------------------------------------
// ret[t, h*16+e] = sum_d q[t,h,d]*kv[b,h,d,e] + x[t, h*16+e]
// ---------------------------------------------------------------------------
__global__ void k_apply(const unsigned short* __restrict__ qkv, const float* __restrict__ kv,
                        const unsigned short* __restrict__ x, unsigned short* __restrict__ ret) {
    int t = blockIdx.x * blockDim.x + threadIdx.x;
    if (t >= NTOK * 16) return;
    int tok = t >> 4, hh = t & 15, b = tok >> 16;
    const unsigned short* q = qkv + (size_t)tok * QKVC + hh * 48;
    const float* kvh = kv + (size_t)(b * 16 + hh) * 256;
    float qv[16];
    #pragma unroll
    for (int j = 0; j < 16; j++) qv[j] = bf2f(q[j]);
    #pragma unroll
    for (int e = 0; e < 16; e++) {
        float a = 0.f;
        #pragma unroll
        for (int d2 = 0; d2 < 16; d2++) a = fmaf(qv[d2], kvh[d2 * 16 + e], a);
        size_t off = (size_t)tok * WIDTHC + hh * 16 + e;
        ret[off] = f2bf(a + bf2f(x[off]));
    }
}

// ---------------------------------------------------------------------------
// Final: fc2 (256->2) + bilinear-border sample of inp, write fp32 output
// ---------------------------------------------------------------------------
__global__ void k_final(const unsigned short* __restrict__ f1, const float* __restrict__ w2,
                        const float* __restrict__ b2, const float* __restrict__ inp,
                        const float* __restrict__ coord, float* __restrict__ out) {
    int t = blockIdx.x * blockDim.x + threadIdx.x;
    if (t >= NTOK) return;
    int b = t >> 16, qy = (t >> 8) & 255, qx = t & 255;
    const unsigned short* p = f1 + (size_t)t * WIDTHC;
    float a0 = b2[0], a1 = b2[1];
    for (int c = 0; c < WIDTHC; c++) {
        float v = bf2f(p[c]);
        a0 = fmaf(w2[c], v, a0);
        a1 = fmaf(w2[WIDTHC + c], v, a1);
    }
    float gy = coord[((size_t)(b * HQ + qy) * HQ + qx) * 2 + 0];
    float gx = coord[((size_t)(b * HQ + qy) * HQ + qx) * 2 + 1];
    float fx = fminf(fmaxf(((gx + 1.f) * 64.f - 1.f) * 0.5f, 0.f), 63.f);
    float fy = fminf(fmaxf(((gy + 1.f) * 64.f - 1.f) * 0.5f, 0.f), 63.f);
    int x0 = (int)floorf(fx), y0 = (int)floorf(fy);
    float wx = fx - (float)x0, wy = fy - (float)y0;
    int x1 = min(x0 + 1, 63), y1 = min(y0 + 1, 63);
    #pragma unroll
    for (int o = 0; o < 2; o++) {
        const float* im = inp + (size_t)(b * 2 + o) * 4096;
        float v00 = im[y0 * 64 + x0], v01 = im[y0 * 64 + x1];
        float v10 = im[y1 * 64 + x0], v11 = im[y1 * 64 + x1];
        float bl = v00 * (1.f - wx) * (1.f - wy) + v01 * wx * (1.f - wy) +
                   v10 * (1.f - wx) * wy + v11 * wx * wy;
        out[((size_t)(b * 2 + o) * HQ + qy) * HQ + qx] = (o ? a1 : a0) + bl;
    }
}

// ---------------------------------------------------------------------------
// Host orchestration
// ---------------------------------------------------------------------------
static inline size_t alignup(size_t x) { return (x + 255) & ~(size_t)255; }

extern "C" void kernel_launch(void* const* d_in, const int* in_sizes, int n_in,
                              void* d_out, int out_size, void* d_ws, size_t ws_size,
                              hipStream_t stream) {
    (void)in_sizes; (void)n_in; (void)out_size; (void)ws_size;
    const float* inp   = (const float*)d_in[0];
    const float* coord = (const float*)d_in[1];
    const float* cell  = (const float*)d_in[2];
    const float* enc_w = (const float*)d_in[3];
    const float* enc_b = (const float*)d_in[4];
    const float* c00_w = (const float*)d_in[5];
    const float* c00_b = (const float*)d_in[6];
    const float* fc1_w = (const float*)d_in[27];
    const float* fc1_b = (const float*)d_in[28];
    const float* fc2_w = (const float*)d_in[29];
    const float* fc2_b = (const float*)d_in[30];

    char* ws = (char*)d_ws;
    size_t off = 0;
    auto take = [&](size_t bytes) { char* p = ws + off; off += alignup(bytes); return p; };

    float*          feat = (float*)take((size_t)2 * ENCC * 4096 * 4);
    unsigned short* grid = (unsigned short*)take((size_t)NTOK * KPAD * 2);
    unsigned short* X    = (unsigned short*)take((size_t)NTOK * WIDTHC * 2);
    unsigned short* QKV  = (unsigned short*)take((size_t)NTOK * QKVC * 2);
    unsigned short* T    = (unsigned short*)take((size_t)NTOK * WIDTHC * 2);
    float*          kv   = (float*)take((size_t)2 * 16 * 256 * 4);
    unsigned short* wc00 = (unsigned short*)take((size_t)256 * KPAD * 2);
    unsigned short *wqkv[2], *wo1[2], *wo2[2];
    for (int i = 0; i < 2; i++) {
        wqkv[i] = (unsigned short*)take((size_t)768 * 256 * 2);
        wo1[i]  = (unsigned short*)take((size_t)256 * 256 * 2);
        wo2[i]  = (unsigned short*)take((size_t)256 * 256 * 2);
    }
    unsigned short* wfc1 = (unsigned short*)take((size_t)256 * 256 * 2);
    unsigned short* Hbuf = QKV;   // reuse: qkv dead once H is built

    auto cvt = [&](const float* s, unsigned short* d, int N, int Ks, int Kd) {
        int n = N * Kd;
        k_cvt_w<<<(n + 255) / 256, 256, 0, stream>>>(s, d, N, Ks, Kd);
    };
    cvt(c00_w, wc00, 256, 266, KPAD);
    for (int i = 0; i < 2; i++) {
        int base = 7 + i * 10;
        cvt((const float*)d_in[base + 0], wqkv[i], 768, 256, 256);
        cvt((const float*)d_in[base + 2], wo1[i],  256, 256, 256);
        cvt((const float*)d_in[base + 4], wo2[i],  256, 256, 256);
    }
    cvt(fc1_w, wfc1, 256, 256, 256);

    k_enc<<<(2 * ENCC * 4096 + 255) / 256, 256, 0, stream>>>(inp, enc_w, enc_b, feat);
    k_build_grid<<<(NTOK + 255) / 256, 256, 0, stream>>>(feat, coord, cell, grid);

    const int M = NTOK;
    dim3 blk(256);
    k_gemm_bf16<0><<<dim3(256 / TN, M / TM), blk, 0, stream>>>(grid, wc00, c00_b, nullptr, X, M, 256, KPAD);

    for (int i = 0; i < 2; i++) {
        int base = 7 + i * 10;
        const float* qkv_b = (const float*)d_in[base + 1];
        const float* o1_b  = (const float*)d_in[base + 3];
        const float* o2_b  = (const float*)d_in[base + 5];
        const float* klnw  = (const float*)d_in[base + 6];
        const float* klnb  = (const float*)d_in[base + 7];
        const float* vlnw  = (const float*)d_in[base + 8];
        const float* vlnb  = (const float*)d_in[base + 9];

        k_gemm_bf16<0><<<dim3(QKVC / TN, M / TM), blk, 0, stream>>>(X, wqkv[i], qkv_b, nullptr, QKV, M, QKVC, 256);
        k_ln_kv<<<(NTOK * 16 + 255) / 256, 256, 0, stream>>>(QKV, klnw, klnb, vlnw, vlnb);
        k_kv<<<32, 256, 0, stream>>>(QKV, kv);
        k_apply<<<(NTOK * 16 + 255) / 256, 256, 0, stream>>>(QKV, kv, X, T);
        k_gemm_bf16<1><<<dim3(256 / TN, M / TM), blk, 0, stream>>>(T, wo1[i], o1_b, nullptr, Hbuf, M, 256, 256);
        k_gemm_bf16<2><<<dim3(256 / TN, M / TM), blk, 0, stream>>>(Hbuf, wo2[i], o2_b, X, X, M, 256, 256);
    }

    k_gemm_bf16<1><<<dim3(256 / TN, M / TM), blk, 0, stream>>>(X, wfc1, fc1_b, nullptr, T, M, 256, 256);
    k_final<<<(NTOK + 255) / 256, 256, 0, stream>>>(T, fc2_w, fc2_b, inp, coord, (float*)d_out);
}
